// BertSelfAttention_65231963292390
// MI455X (gfx1250) — matrix operand order repaired and checked by reading, not compiled, not run on hardware
//
#include <hip/hip_runtime.h>
#include <hip/hip_bf16.h>
#include <cstdint>

#define B_    2
#define S_    1024
#define H_    1024
#define NH_   16
#define D_    64
#define BINS_ 32

typedef __attribute__((ext_vector_type(16))) __bf16 v16bf;
typedef __attribute__((ext_vector_type(8)))  __bf16 v8bf;
typedef __attribute__((ext_vector_type(8)))  float  v8f;
typedef __attribute__((ext_vector_type(4)))  float  v4f;
typedef __bf16 bft;

static __device__ __forceinline__ v8f wmma_bf16(v16bf a, v16bf b, v8f c) {
  return __builtin_amdgcn_wmma_f32_16x16x32_bf16(false, a, false, b, (short)0, c,
                                                 false, false);
}

// 16B-per-lane async copy global -> LDS (ASYNCcnt-tracked, CDNA5)
static __device__ __forceinline__ void async_ld16(uint32_t lds, const void* g) {
  asm volatile("global_load_async_to_lds_b128 %0, %1, off"
               :: "v"(lds), "v"(g) : "memory");
}

// ---------------------------------------------------------------- f32 -> bf16
__global__ void cvt_bf16_kernel(const float* __restrict__ in,
                                bft* __restrict__ out, int n) {
  int i = (blockIdx.x * blockDim.x + threadIdx.x) * 4;
  if (i + 3 < n) {
    v4f f = *(const v4f*)(in + i);
    out[i + 0] = (bft)f.x; out[i + 1] = (bft)f.y;
    out[i + 2] = (bft)f.z; out[i + 3] = (bft)f.w;
  }
}

// ---------------------------------------------- one-hot [B,S,S,32] -> u8 index
// 256 MB read-once stream: use NT loads so it does not evict the L2-resident
// Q/K/V working set.
__global__ void binidx_kernel(const float* __restrict__ rel_pos,
                              uint8_t* __restrict__ idx) {
  size_t i = (size_t)blockIdx.x * blockDim.x + threadIdx.x;  // over B*S*S
  const v4f* p = (const v4f*)(rel_pos + i * BINS_);
  int best = 0;
#pragma unroll
  for (int q = 0; q < 8; ++q) {
    v4f f = __builtin_nontemporal_load(p + q);
    if (f.x > 0.5f) best = 4 * q + 0;
    if (f.y > 0.5f) best = 4 * q + 1;
    if (f.z > 0.5f) best = 4 * q + 2;
    if (f.w > 0.5f) best = 4 * q + 3;
  }
  idx[i] = (uint8_t)best;
}

// ----------------------------- unary[b,h,k] = dot(hidden[b,k,:], kuw[h,:]) ---
__global__ void unary_kernel(const float* __restrict__ hid,
                             const float* __restrict__ kuw,
                             float* __restrict__ unary) {
  int lane = threadIdx.x & 31;
  int wid  = blockIdx.x * (blockDim.x >> 5) + (threadIdx.x >> 5);  // B*NH*S
  int k  = wid % S_;
  int bh = wid / S_;
  int h  = bh % NH_;
  int bb = bh / NH_;
  const float* hp = hid + ((size_t)bb * S_ + k) * H_;
  const float* wp = kuw + (size_t)h * H_;
  float acc = 0.f;
  for (int c = lane; c < H_; c += 32) acc += hp[c] * wp[c];
#pragma unroll
  for (int s = 1; s < 32; s <<= 1) acc += __shfl_xor(acc, s, 32);
  if (lane == 0) unary[wid] = acc;
}

// --------------------------------------------------- QKV GEMM (bf16 WMMA) ---
// C[m,n] = sum_k hid[m,k]*W[n,k];  M=2048, N=3072, K=1024.
// One wave owns a 16x64 tile (A reused across 4 B subtiles).
__global__ void qkv_wmma_kernel(const bft* __restrict__ hid,
                                const bft* __restrict__ wt,
                                const float* __restrict__ q_bias,
                                const float* __restrict__ v_bias,
                                bft* __restrict__ Qo, bft* __restrict__ Ko,
                                bft* __restrict__ VTo) {
  const int lane = threadIdx.x & 31;
  const int wid  = blockIdx.x * (blockDim.x >> 5) + (threadIdx.x >> 5);
  const int NTN  = (3 * H_) / 64;          // 48 tiles of 64 cols
  const int tm   = wid / NTN;
  const int tn   = wid % NTN;
  const int mbase = tm * 16, nbase = tn * 64;
  const int hl = lane >> 4, l16 = lane & 15;

  v8f acc[4] = {};
  const bft* arow = hid + (size_t)(mbase + l16) * H_;
  const int abnd = hl * 16;   // A K-band: lanes 0-15 -> K 0..15, hi -> 16..31
  const int bbnd = hl * 16;   // B K-band: lanes 0-15 -> K 0..15, hi -> 16..31

  for (int k0 = 0; k0 < H_; k0 += 32) {
    union { v16bf v; v8bf h[2]; } ua;
    ua.h[0] = *(const v8bf*)(arow + k0 + abnd);
    ua.h[1] = *(const v8bf*)(arow + k0 + abnd + 8);
    __builtin_prefetch(arow + k0 + 64, 0, 0);
#pragma unroll
    for (int t = 0; t < 4; ++t) {
      const int n = nbase + t * 16 + l16;
      v16bf b = *(const v16bf*)(wt + (size_t)n * H_ + k0 + bbnd);
      acc[t] = wmma_bf16(ua.v, b, acc[t]);
    }
  }

  const float rsd = 0.125f;  // 1/sqrt(D)
#pragma unroll
  for (int t = 0; t < 4; ++t) {
#pragma unroll
    for (int j = 0; j < 8; ++j) {
      int m = mbase + j + 8 * hl;          // C/D layout: row = VGPR + 8*half
      int n = nbase + t * 16 + l16;        // col = lane%16
      int bb = m / S_, srow = m % S_;
      float c = acc[t][j];
      if (n < H_) {
        int h = n / D_, d = n % D_;
        Qo[(((size_t)bb * NH_ + h) * S_ + srow) * D_ + d] =
            (bft)((c + q_bias[n]) * rsd);
      } else if (n < 2 * H_) {
        int nn = n - H_;
        int h = nn / D_, d = nn % D_;
        Ko[(((size_t)bb * NH_ + h) * S_ + srow) * D_ + d] = (bft)c;
      } else {
        int nn = n - 2 * H_;
        int h = nn / D_, d = nn % D_;
        VTo[(((size_t)bb * NH_ + h) * D_ + d) * S_ + srow] =
            (bft)(c + v_bias[nn]);
      }
    }
  }
}

// ------------------------------- fused flash attention ----------------------
// One block = one (b,h) x 128 query rows (8 waves x 16-row q-tiles).
// K/V 32-key chunks (8 KB) are async-copied to LDS once per BLOCK and
// double-buffered: next chunk streams in (ASYNCcnt) while the current chunk's
// 8 WMMAs execute. Each wave issues exactly 2 async ops per chunk, so
// s_wait_asynccnt 0x2 == "current chunk resident" (async loads complete
// in order).
__global__ void attn_kernel(const bft* __restrict__ Q, const bft* __restrict__ K,
                            const bft* __restrict__ VT,
                            const uint8_t* __restrict__ mask,
                            const uint8_t* __restrict__ binidx,
                            const float* __restrict__ rpw,
                            const float* __restrict__ unary,
                            float* __restrict__ out) {
  __shared__ float s_rpw[NH_ * BINS_];
  __shared__ __align__(64) bft s_p[8][16 * 32];  // per-wave P bounce (1 KB)
  __shared__ __align__(64) bft s_kv[2][4096];    // [buf][K: 32x64 | V: 64x32]

  const int lane = threadIdx.x & 31;
  const int w    = threadIdx.x >> 5;
  const int bh   = blockIdx.x >> 3;              // 0..31  (b*NH + h)
  const int qg   = blockIdx.x & 7;               // 8 q-groups of 128 rows
  const int qt   = qg * 8 + w;                   // this wave's 16-row q-tile
  const int bb = bh / NH_, h = bh % NH_;
  const int hl = lane >> 4, l16 = lane & 15;

  for (int i = threadIdx.x; i < NH_ * BINS_; i += blockDim.x) s_rpw[i] = rpw[i];
  const float* rpw_h = &s_rpw[h * BINS_];

  // ---- stage helper: 8 KB chunk, 512 x 16B segments over 256 threads -------
  const char* kvK = (const char*)(K + (size_t)bh * S_ * D_);   // + kb*128
  const char* kvV = (const char*)VT + (size_t)bh * D_ * S_ * 2;  // + d*2048+kb*2
#define STAGE_CHUNK(kbn, bufp)                                                 \
  {                                                                            \
    uint32_t lbase = (uint32_t)(uintptr_t)(&s_kv[(bufp)][0]);                  \
    _Pragma("unroll") for (int i_ = 0; i_ < 2; ++i_) {                         \
      int o_ = (i_ * 256 + (int)threadIdx.x) * 16;                             \
      const char* g_;                                                          \
      if (o_ < 4096)  /* K slice: rows contiguous -> 4KB linear */             \
        g_ = kvK + (size_t)(kbn)*128 + o_;                                     \
      else {                                                                   \
        int o2_ = o_ - 4096; /* V slice: row d (64B) stride 2048B */           \
        g_ = kvV + (size_t)(o2_ >> 6) * (S_ * 2) + (size_t)(kbn)*2 +           \
             (o2_ & 63);                                                       \
      }                                                                        \
      async_ld16(lbase + o_, g_);                                              \
    }                                                                          \
  }

  // Q A-matrices (d 0..31 and 32..63), loaded once per wave
  const bft* qp = Q + ((size_t)bh * S_ + (qt * 16 + l16)) * D_;
  union { v16bf v; v8bf hh[2]; } uq0, uq1;
  uq0.hh[0] = *(const v8bf*)(qp + hl * 16);
  uq0.hh[1] = *(const v8bf*)(qp + hl * 16 + 8);
  uq1.hh[0] = *(const v8bf*)(qp + 32 + hl * 16);
  uq1.hh[1] = *(const v8bf*)(qp + 32 + hl * 16 + 8);

  float mrun[8], lrun[8];
#pragma unroll
  for (int j = 0; j < 8; ++j) { mrun[j] = -1e30f; lrun[j] = 0.f; }
  v8f co[4] = {};

  const uint8_t* idxb = binidx + (size_t)bb * S_ * S_;
  const float*   un   = unary + (size_t)bh * S_;
  const uint8_t* mk   = mask + (size_t)bb * S_;

  STAGE_CHUNK(0, 0);   // prologue: chunk 0 in flight

  int buf = 0;
  for (int kb = 0; kb < S_; kb += 32, buf ^= 1) {
    if (kb + 32 < S_) {            // uniform across block
      STAGE_CHUNK(kb + 32, buf ^ 1);
      asm volatile("s_wait_asynccnt 0x2" ::: "memory");
    } else {
      asm volatile("s_wait_asynccnt 0x0" ::: "memory");
    }
    __syncthreads();               // whole 8 KB chunk resident for all waves
    const bft* kvb = &s_kv[buf][0];

    float sc[2][8];
#pragma unroll
    for (int t = 0; t < 2; ++t) {
      const int kl = t * 16 + l16;                 // key within chunk
      const int kc = kb + kl;                      // global key
      v16bf b0 = *(const v16bf*)(kvb + (size_t)kl * 64 + hl * 16);       // d 0..31
      v16bf b1 = *(const v16bf*)(kvb + (size_t)kl * 64 + 32 + hl * 16);  // d 32..63
      v8f c = {};
      c = wmma_bf16(uq0.v, b0, c);
      c = wmma_bf16(uq1.v, b1, c);
      const float uk = un[kc];
      const bool  mm = mk[kc] != 0;
#pragma unroll
      for (int j = 0; j < 8; ++j) {
        int q = qt * 16 + j + 8 * hl;
        float v = c[j] + uk + rpw_h[idxb[(size_t)q * S_ + kc]];
        sc[t][j] = mm ? -1e30f : v;
      }
    }
    // online softmax: row reductions stay inside 16-lane halves
    float alpha[8];
#pragma unroll
    for (int j = 0; j < 8; ++j) {
      float mx = fmaxf(sc[0][j], sc[1][j]);
#pragma unroll
      for (int s = 1; s < 16; s <<= 1) mx = fmaxf(mx, __shfl_xor(mx, s, 32));
      float mn = fmaxf(mrun[j], mx);
      alpha[j] = __expf(mrun[j] - mn);
      mrun[j]  = mn;
      float p0 = __expf(sc[0][j] - mn);
      float p1 = __expf(sc[1][j] - mn);
      sc[0][j] = p0; sc[1][j] = p1;
      float rs = p0 + p1;
#pragma unroll
      for (int s = 1; s < 16; s <<= 1) rs += __shfl_xor(rs, s, 32);
      lrun[j] = lrun[j] * alpha[j] + rs;
    }
#pragma unroll
    for (int dt = 0; dt < 4; ++dt)
#pragma unroll
      for (int j = 0; j < 8; ++j) co[dt][j] = co[dt][j] * alpha[j];

    // C/D-layout P -> row-major LDS -> A-layout reload (lane transpose)
    bft* pp = &s_p[w][0];
#pragma unroll
    for (int t = 0; t < 2; ++t)
#pragma unroll
      for (int j = 0; j < 8; ++j)
        pp[(j + 8 * hl) * 32 + t * 16 + l16] = (bft)sc[t][j];
    asm volatile("s_wait_dscnt 0x0" ::: "memory");
    union { v16bf v; v8bf hh[2]; } up;
    const bft* prow = pp + l16 * 32;
    up.hh[0] = *(const v8bf*)(prow + hl * 16);
    up.hh[1] = *(const v8bf*)(prow + hl * 16 + 8);

#pragma unroll
    for (int dt = 0; dt < 4; ++dt) {
      const int d = dt * 16 + l16;
      v16bf bv = *(const v16bf*)(kvb + 2048 + (size_t)d * 32 + hl * 16);
      co[dt] = wmma_bf16(up.v, bv, co[dt]);
    }
    __syncthreads();               // all waves done with buf before overwrite
  }

#pragma unroll
  for (int dt = 0; dt < 4; ++dt)
#pragma unroll
    for (int j = 0; j < 8; ++j) {
      int q = qt * 16 + j + 8 * hl;
      int d = dt * 16 + l16;
      out[((size_t)bb * S_ + q) * H_ + h * D_ + d] = co[dt][j] / lrun[j];
    }
#undef STAGE_CHUNK
}

// ---------------------------------------------------------------------------
extern "C" void kernel_launch(void* const* d_in, const int* in_sizes, int n_in,
                              void* d_out, int out_size, void* d_ws,
                              size_t ws_size, hipStream_t stream) {
  (void)in_sizes; (void)n_in; (void)out_size; (void)ws_size;
  const float*   hidden = (const float*)d_in[0];
  const uint8_t* maskp  = (const uint8_t*)d_in[1];   // jnp bool -> 1 byte
  const float*   qkv_w  = (const float*)d_in[2];
  const float*   q_bias = (const float*)d_in[3];
  const float*   v_bias = (const float*)d_in[4];
  const float*   relpos = (const float*)d_in[5];
  const float*   rpw    = (const float*)d_in[6];
  const float*   kuw    = (const float*)d_in[7];
  float* out = (float*)d_out;

  char* p = (char*)d_ws;
  auto take = [&](size_t bytes) {
    char* r = p;
    p += (bytes + 255) & ~(size_t)255;
    return r;
  };
  bft*     hid_bf = (bft*)take((size_t)B_ * S_ * H_ * 2);
  bft*     w_bf   = (bft*)take((size_t)3 * H_ * H_ * 2);
  bft*     Qb     = (bft*)take((size_t)B_ * NH_ * S_ * D_ * 2);
  bft*     Kb     = (bft*)take((size_t)B_ * NH_ * S_ * D_ * 2);
  bft*     VTb    = (bft*)take((size_t)B_ * NH_ * S_ * D_ * 2);
  uint8_t* idxb   = (uint8_t*)take((size_t)B_ * S_ * S_);
  float*   un     = (float*)take((size_t)B_ * NH_ * S_ * 4);

  cvt_bf16_kernel<<<(B_ * S_ * H_ / 4) / 256, 256, 0, stream>>>(
      hidden, hid_bf, B_ * S_ * H_);
  cvt_bf16_kernel<<<(3 * H_ * H_ / 4) / 256, 256, 0, stream>>>(
      qkv_w, w_bf, 3 * H_ * H_);
  binidx_kernel<<<(B_ * S_ * S_) / 256, 256, 0, stream>>>(relpos, idxb);
  unary_kernel<<<(B_ * NH_ * S_) / 8, 256, 0, stream>>>(hidden, kuw, un);
  qkv_wmma_kernel<<<(128 * 48) / 8, 256, 0, stream>>>(hid_bf, w_bf, q_bias,
                                                      v_bias, Qb, Kb, VTb);
  attn_kernel<<<B_ * NH_ * (S_ / 128), 256, 0, stream>>>(
      Qb, Kb, VTb, maskp, idxb, rpw, un, out);
}
